// IterativeSceneGraphGeneration_4870492913932
// MI455X (gfx1250) — matrix operand order repaired and checked by reading, not compiled, not run on hardware
//
#include <hip/hip_runtime.h>
#include <hip/hip_bf16.h>

// ---------- sizes ----------
#define EMBED   512
#define MID     512
#define FINAL   1024
#define N_OBJ   2048
#define N_REL   32768
#define NUM_OBJ 151
#define NUM_REL 51
#define KPAD    160   // 151 padded to multiple of 32 for bf16 WMMA K-steps

typedef __attribute__((ext_vector_type(16))) __bf16 v16bf;
typedef __attribute__((ext_vector_type(8)))  float  v8f;

union Frag16 { v16bf v; uint4 q[2]; };

__device__ __forceinline__ unsigned short f2bf(float x) {
  union { float f; unsigned int u; } v; v.f = x;
  unsigned int r = v.u + 0x7FFFu + ((v.u >> 16) & 1u);   // round-to-nearest-even
  return (unsigned short)(r >> 16);
}

// ---------- utility ----------
__global__ void zero_f(float* p, int n) {
  int i = blockIdx.x * 256 + threadIdx.x;
  if (i < n) p[i] = 0.f;
}

// ---------- pass 1: per-object row sums of graph -> inv[o] = 1/(sum+1e-9) ----------
__global__ __launch_bounds__(256) void rowsum_inv(const float* __restrict__ graph,
                                                  float* __restrict__ inv) {
  __shared__ float red[256];
  const int o = blockIdx.x;
  const float4* row = (const float4*)(graph + (size_t)o * N_REL);
  float s = 0.f;
  for (int i = threadIdx.x; i < N_REL / 4; i += 256) {
    float4 v = row[i];
    s += v.x + v.y + v.z + v.w;
  }
  red[threadIdx.x] = s; __syncthreads();
  for (int st = 128; st > 0; st >>= 1) {
    if (threadIdx.x < st) red[threadIdx.x] += red[threadIdx.x + st];
    __syncthreads();
  }
  if (threadIdx.x == 0) inv[o] = 1.0f / (red[0] + 1e-9f);
}

// ---------- pass 2: Gk[r,k] = sum_{o: ent[o]==k} inv[o]*graph[o,r]  (bf16, K padded) ----------
__global__ __launch_bounds__(64) void build_gk(const float* __restrict__ graph,
                                               const int* __restrict__ ents,
                                               const float* __restrict__ inv,
                                               unsigned short* __restrict__ Gk) {
  __shared__ float acc[64][153];     // pitch 153 (odd) -> conflict-free per-lane rows
  __shared__ int   se[N_OBJ];
  __shared__ float si[N_OBJ];
  const int t = threadIdx.x;
  for (int j = t; j < N_OBJ; j += 64) { se[j] = ents[j]; si[j] = inv[j]; }
  for (int k = 0; k < 153; ++k) acc[t][k] = 0.f;
  __syncthreads();
  const int rbase = blockIdx.x * 64;
  const float* gcol = graph + rbase + t;
  #pragma unroll 4
  for (int o = 0; o < N_OBJ; ++o) {
    float v = gcol[(size_t)o * N_REL];      // 64 consecutive floats per block per o
    acc[t][se[o]] += v * si[o];
  }
  unsigned short* out = Gk + (size_t)(rbase + t) * KPAD;
  for (int k = 0; k < NUM_OBJ; ++k) out[k] = f2bf(acc[t][k]);
  for (int k = NUM_OBJ; k < KPAD; ++k) out[k] = 0;
}

// ---------- deterministic Frobenius sum-of-squares (two-stage) ----------
__global__ __launch_bounds__(256) void sumsq_part(const float* __restrict__ x, int n,
                                                  float* __restrict__ part) {
  __shared__ float red[256];
  float s = 0.f;
  for (int i = blockIdx.x * 256 + threadIdx.x; i < n; i += gridDim.x * 256) {
    float v = x[i]; s += v * v;
  }
  red[threadIdx.x] = s; __syncthreads();
  for (int st = 128; st > 0; st >>= 1) {
    if (threadIdx.x < st) red[threadIdx.x] += red[threadIdx.x + st];
    __syncthreads();
  }
  if (threadIdx.x == 0) part[blockIdx.x] = red[0];
}
__global__ __launch_bounds__(256) void reduce_part(const float* __restrict__ part,
                                                   float* __restrict__ out) {
  __shared__ float red[256];
  red[threadIdx.x] = part[blockIdx.x * 256 + threadIdx.x];
  __syncthreads();
  for (int st = 128; st > 0; st >>= 1) {
    if (threadIdx.x < st) red[threadIdx.x] += red[threadIdx.x + st];
    __syncthreads();
  }
  if (threadIdx.x == 0) out[blockIdx.x] = red[0];
}

// ---------- small fp32 GEMM: out[k,m] = dot(tab[k,0:512], W[m, off:off+512]) ----------
__global__ __launch_bounds__(128) void tabmm(const float* __restrict__ tab,
                                             const float* __restrict__ W,
                                             int ldw, int off,
                                             float* __restrict__ out) {
  __shared__ float trow[EMBED];
  const int k = blockIdx.y;
  const int m = blockIdx.x * 128 + threadIdx.x;
  const float* tr = tab + (size_t)k * EMBED;
  for (int j = threadIdx.x; j < EMBED; j += 128) trow[j] = tr[j];
  __syncthreads();
  const float4* wr = (const float4*)(W + (size_t)m * ldw + off);
  const float4* tv = (const float4*)trow;
  float s = 0.f;
  for (int j = 0; j < EMBED / 4; ++j) {
    float4 a = tv[j], b = wr[j];
    s += a.x * b.x + a.y * b.y + a.z * b.z + a.w * b.w;
  }
  out[(size_t)k * MID + m] = s;
}

// ---------- cq[m] = dot(c, Wq[m,:]) ----------
__global__ __launch_bounds__(256) void vecmat(const float* __restrict__ vec,
                                              const float* __restrict__ W,
                                              float* __restrict__ out) {
  __shared__ float vs[EMBED];
  const int m = blockIdx.x * 256 + threadIdx.x;
  for (int j = threadIdx.x; j < EMBED; j += 256) vs[j] = vec[j];
  __syncthreads();
  const float* wr = W + (size_t)m * EMBED;
  float s = 0.f;
  for (int j = 0; j < EMBED; ++j) s += vs[j] * wr[j];
  out[m] = s;
}

// ---------- qk_bf16[k,m] = bf16(relu(sq*(EQ[k,m]+cq[m]) + bq[m])), rows >=151 zero ----------
__global__ __launch_bounds__(256) void make_qk(const float* __restrict__ EQ,
                                               const float* __restrict__ cq,
                                               const float* __restrict__ bq,
                                               const float* __restrict__ qg,
                                               const float* __restrict__ ssq,
                                               unsigned short* __restrict__ qk) {
  const int idx = blockIdx.x * 256 + threadIdx.x;  // KPAD*MID
  const int k = idx >> 9, m = idx & 511;
  unsigned short r = 0;
  if (k < NUM_OBJ) {
    float sq = qg[0] / sqrtf(ssq[0]);
    float v = sq * (EQ[(size_t)k * MID + m] + cq[m]) + bq[m];
    r = f2bf(fmaxf(v, 0.f));
  }
  qk[(size_t)m * KPAD + k] = r;
}

// ---------- main WMMA GEMM: A_[M,512] = Gk[M,160](bf16) @ qk[160,512](bf16), f32 acc ----------
// Block = 256 threads = 8 waves. Block tile 128(M) x 64(N); wave: 16(M) x 4x16(N).
__global__ __launch_bounds__(256) void gemm_bf16_wmma(const unsigned short* __restrict__ A,
                                                      const unsigned short* __restrict__ B,
                                                      float* __restrict__ C) {
  const int lane  = threadIdx.x & 31;
  const int wave  = threadIdx.x >> 5;
  const int rbase = blockIdx.y * 128 + wave * 16;
  const int nbase = blockIdx.x * 64;
  const int mrow  = lane & 15;
  const int khalf = (lane >> 4) & 1;       // ISA A-layout: lanes 16-31 hold K-offset +8

  v8f acc0 = {}, acc1 = {}, acc2 = {}, acc3 = {};

  #pragma unroll
  for (int ks = 0; ks < KPAD / 32; ++ks) {
    Frag16 a;
    const unsigned short* ap = A + (size_t)(rbase + mrow) * KPAD + ks * 32 + khalf * 8;
    a.q[0] = *(const uint4*)(ap);          // comps 0-7  : K = ks*32 + khalf*8 + [0,8)
    a.q[1] = *(const uint4*)(ap + 16);     // comps 8-15 : K = ks*32 + khalf*8 + 16 + [0,8)

    const unsigned short* bp = B + (size_t)(nbase + mrow) * KPAD + ks * 32 + khalf * 8;  // lane&15 = N col, K as A
    Frag16 b0, b1, b2, b3;
    b0.q[0] = *(const uint4*)(bp +  0 * KPAD); b0.q[1] = *(const uint4*)(bp +  0 * KPAD + 16);
    b1.q[0] = *(const uint4*)(bp + 16 * KPAD); b1.q[1] = *(const uint4*)(bp + 16 * KPAD + 16);
    b2.q[0] = *(const uint4*)(bp + 32 * KPAD); b2.q[1] = *(const uint4*)(bp + 32 * KPAD + 16);
    b3.q[0] = *(const uint4*)(bp + 48 * KPAD); b3.q[1] = *(const uint4*)(bp + 48 * KPAD + 16);

    acc0 = __builtin_amdgcn_wmma_f32_16x16x32_bf16(false, a.v, false, b0.v, (short)0, acc0, false, false);
    acc1 = __builtin_amdgcn_wmma_f32_16x16x32_bf16(false, a.v, false, b1.v, (short)0, acc1, false, false);
    acc2 = __builtin_amdgcn_wmma_f32_16x16x32_bf16(false, a.v, false, b2.v, (short)0, acc2, false, false);
    acc3 = __builtin_amdgcn_wmma_f32_16x16x32_bf16(false, a.v, false, b3.v, (short)0, acc3, false, false);
  }

  // D layout: comp v -> M = v (lanes 0-15) or 8+v (lanes 16-31); N = lane&15
  const int col  = lane & 15;
  const int rofs = (lane < 16) ? 0 : 8;
  float* crow = C + (size_t)(rbase + rofs) * MID + nbase + col;
  #pragma unroll
  for (int v = 0; v < 8; ++v) {
    crow[(size_t)v * MID +  0] = acc0[v];
    crow[(size_t)v * MID + 16] = acc1[v];
    crow[(size_t)v * MID + 32] = acc2[v];
    crow[(size_t)v * MID + 48] = acc3[v];
  }
}

// ---------- h partials: acc over 256 relations of relu(sv*(HW+TW+PW)+bv) * A_ ----------
__global__ __launch_bounds__(256) void h_partial(const float* __restrict__ A_,
                                                 const int* __restrict__ rel,
                                                 const float* __restrict__ HW,
                                                 const float* __restrict__ TW,
                                                 const float* __restrict__ PW,
                                                 const float* __restrict__ bv,
                                                 const float* __restrict__ vg,
                                                 const float* __restrict__ ssv,
                                                 float* __restrict__ partial) {
  const int m  = blockIdx.x * 256 + threadIdx.x;   // 0..511
  const int rb = blockIdx.y;                       // 0..127
  const float sv  = vg[0] / sqrtf(ssv[0]);
  const float bvm = bv[m];
  float acc = 0.f;
  const int r0 = rb * 256;
  for (int rr = 0; rr < 256; ++rr) {
    const int r = r0 + rr;
    const int i0 = rel[3 * r], i1 = rel[3 * r + 1], i2 = rel[3 * r + 2];
    float v = sv * (HW[(size_t)i0 * MID + m] + TW[(size_t)i1 * MID + m]
                  + PW[(size_t)i2 * MID + m]) + bvm;
    acc += fmaxf(v, 0.f) * A_[(size_t)r * MID + m];
  }
  partial[(size_t)rb * MID + m] = acc;
}

__global__ __launch_bounds__(256) void h_reduce(const float* __restrict__ partial,
                                                float* __restrict__ h) {
  const int m = blockIdx.x * 256 + threadIdx.x;
  float s = 0.f;
  for (int c = 0; c < 128; ++c) s += partial[(size_t)c * MID + m];
  h[m] = s;
}

// ---------- c[m] += sa * dot(h, Va[m,:]) + ba[m] ----------
__global__ __launch_bounds__(256) void update_c(const float* __restrict__ h,
                                                const float* __restrict__ Va,
                                                const float* __restrict__ ba,
                                                const float* __restrict__ ag,
                                                const float* __restrict__ ssa,
                                                float* __restrict__ c) {
  __shared__ float hs[MID];
  const int m = blockIdx.x * 256 + threadIdx.x;
  for (int j = threadIdx.x; j < MID; j += 256) hs[j] = h[j];
  __syncthreads();
  const float sa = ag[0] / sqrtf(ssa[0]);
  const float* wr = Va + (size_t)m * MID;
  float s = 0.f;
  for (int j = 0; j < MID; ++j) s += hs[j] * wr[j];
  c[m] += sa * s + ba[m];
}

// ---------- sg[m] = sum_o obj_tab[ent[o],m] + 2048*c[m] ----------
__global__ __launch_bounds__(256) void make_sg(const float* __restrict__ obj_tab,
                                               const int* __restrict__ ents,
                                               const float* __restrict__ c,
                                               float* __restrict__ sg) {
  __shared__ int se[N_OBJ];
  const int m = blockIdx.x * 256 + threadIdx.x;
  for (int j = threadIdx.x; j < N_OBJ; j += 256) se[j] = ents[j];
  __syncthreads();
  float s = 0.f;
  for (int o = 0; o < N_OBJ; ++o) s += obj_tab[(size_t)se[o] * EMBED + m];
  sg[m] = s + (float)N_OBJ * c[m];
}

// ---------- out[m] = relu(dot(vec, W[m,:]) + bias[m]) ----------
__global__ __launch_bounds__(256) void fc_relu(const float* __restrict__ vec,
                                               const float* __restrict__ W,
                                               const float* __restrict__ bias,
                                               float* __restrict__ out) {
  __shared__ float vs[MID];
  const int m = blockIdx.x * 256 + threadIdx.x;
  for (int j = threadIdx.x; j < MID; j += 256) vs[j] = vec[j];
  __syncthreads();
  const float* wr = W + (size_t)m * MID;
  float s = 0.f;
  for (int j = 0; j < MID; ++j) s += vs[j] * wr[j];
  out[m] = fmaxf(s + bias[m], 0.f);
}

extern "C" void kernel_launch(void* const* d_in, const int* in_sizes, int n_in,
                              void* d_out, int out_size, void* d_ws, size_t ws_size,
                              hipStream_t stream) {
  (void)in_sizes; (void)n_in; (void)out_size; (void)ws_size;

  const int*   entities = (const int*)  d_in[0];
  const int*   relations= (const int*)  d_in[1];
  const float* graph    = (const float*)d_in[2];
  const float* obj_tab  = (const float*)d_in[3];
  const float* head_tab = (const float*)d_in[4];
  const float* tail_tab = (const float*)d_in[5];
  const float* pred_tab = (const float*)d_in[6];
  const float* lin_v_v  = (const float*)d_in[7];
  const float* lin_v_g  = (const float*)d_in[8];
  const float* lin_v_b  = (const float*)d_in[9];
  const float* lin_q_v  = (const float*)d_in[10];
  const float* lin_q_g  = (const float*)d_in[11];
  const float* lin_q_b  = (const float*)d_in[12];
  const float* lin_a_v  = (const float*)d_in[13];
  const float* lin_a_g  = (const float*)d_in[14];
  const float* lin_a_b  = (const float*)d_in[15];
  const float* fc1_w    = (const float*)d_in[16];
  const float* fc1_b    = (const float*)d_in[17];
  const float* fc2_w    = (const float*)d_in[18];
  const float* fc2_b    = (const float*)d_in[19];

  // workspace layout (256B-aligned slabs)
  char* ws = (char*)d_ws;
  size_t cur = 0;
  auto take = [&](size_t bytes) { size_t s = cur; cur += (bytes + 255) & ~(size_t)255; return s; };
  float* inv      = (float*)(ws + take(N_OBJ * 4));
  float* sumsq    = (float*)(ws + take(6 * 4));              // [v0,v1,q0,q1,a0,a1]
  float* normpart = (float*)(ws + take(6 * 256 * 4));
  float* cvec     = (float*)(ws + take(MID * 4));
  float* cq       = (float*)(ws + take(MID * 4));
  float* h        = (float*)(ws + take(MID * 4));
  float* sg       = (float*)(ws + take(MID * 4));
  float* o1       = (float*)(ws + take(MID * 4));
  float* EQ       = (float*)(ws + take((size_t)NUM_OBJ * MID * 4));
  float* HW       = (float*)(ws + take((size_t)NUM_OBJ * MID * 4));
  float* TW       = (float*)(ws + take((size_t)NUM_OBJ * MID * 4));
  float* PW       = (float*)(ws + take((size_t)NUM_REL * MID * 4));
  float* partial  = (float*)(ws + take((size_t)128 * MID * 4));
  unsigned short* qk = (unsigned short*)(ws + take((size_t)KPAD * MID * 2));
  unsigned short* Gk = (unsigned short*)(ws + take((size_t)N_REL * KPAD * 2));
  float* A_       = (float*)(ws + take((size_t)N_REL * MID * 4));

  // init accumulator state
  zero_f<<<1, 256, 0, stream>>>(sumsq, 6);
  zero_f<<<2, 256, 0, stream>>>(cvec, MID);

  // graph passes: rowsum inverse, then entity-bucketed aggregate Gk (bf16, padded K)
  rowsum_inv<<<N_OBJ, 256, 0, stream>>>(graph, inv);
  build_gk<<<N_REL / 64, 64, 0, stream>>>(graph, entities, inv, Gk);

  // Frobenius norms (deterministic two-stage)
  sumsq_part<<<256, 256, 0, stream>>>(lin_v_v,                 MID * 3 * EMBED, normpart + 0 * 256);
  sumsq_part<<<256, 256, 0, stream>>>(lin_v_v + MID * 3 * EMBED, MID * 3 * EMBED, normpart + 1 * 256);
  sumsq_part<<<256, 256, 0, stream>>>(lin_q_v,                 MID * EMBED,     normpart + 2 * 256);
  sumsq_part<<<256, 256, 0, stream>>>(lin_q_v + MID * EMBED,   MID * EMBED,     normpart + 3 * 256);
  sumsq_part<<<256, 256, 0, stream>>>(lin_a_v,                 MID * MID,       normpart + 4 * 256);
  sumsq_part<<<256, 256, 0, stream>>>(lin_a_v + MID * MID,     MID * MID,       normpart + 5 * 256);
  reduce_part<<<6, 256, 0, stream>>>(normpart, sumsq);

  for (int g = 0; g < 2; ++g) {
    const float* Wv = lin_v_v + (size_t)g * MID * 3 * EMBED;
    const float* Wq = lin_q_v + (size_t)g * MID * EMBED;
    const float* Wa = lin_a_v + (size_t)g * MID * MID;

    tabmm<<<dim3(4, NUM_OBJ), 128, 0, stream>>>(obj_tab,  Wq, EMBED,     0,    EQ);
    tabmm<<<dim3(4, NUM_OBJ), 128, 0, stream>>>(head_tab, Wv, 3 * EMBED, 0,    HW);
    tabmm<<<dim3(4, NUM_OBJ), 128, 0, stream>>>(tail_tab, Wv, 3 * EMBED, 512,  TW);
    tabmm<<<dim3(4, NUM_REL), 128, 0, stream>>>(pred_tab, Wv, 3 * EMBED, 1024, PW);

    vecmat<<<2, 256, 0, stream>>>(cvec, Wq, cq);
    make_qk<<<(KPAD * MID) / 256, 256, 0, stream>>>(EQ, cq, lin_q_b + g * MID,
                                                    lin_q_g + g, sumsq + 2 + g, qk);

    // A_[32768,512] = Gk @ qk  (bf16 WMMA, f32 accumulate)
    gemm_bf16_wmma<<<dim3(MID / 64, N_REL / 128), 256, 0, stream>>>(Gk, qk, A_);

    h_partial<<<dim3(2, 128), 256, 0, stream>>>(A_, relations, HW, TW, PW,
                                                lin_v_b + g * MID, lin_v_g + g,
                                                sumsq + g, partial);
    h_reduce<<<2, 256, 0, stream>>>(partial, h);
    update_c<<<2, 256, 0, stream>>>(h, Wa, lin_a_b + g * MID, lin_a_g + g,
                                    sumsq + 4 + g, cvec);
  }

  make_sg<<<2, 256, 0, stream>>>(obj_tab, entities, cvec, sg);
  fc_relu<<<2, 256, 0, stream>>>(sg, fc1_w, fc1_b, o1);
  fc_relu<<<4, 256, 0, stream>>>(o1, fc2_w, fc2_b, (float*)d_out);
}
